// QNN_03_78778290143344
// MI455X (gfx1250) — matrix operand order repaired and checked by reading, not compiled, not run on hardware
//
#include <hip/hip_runtime.h>
#include <hip/hip_bf16.h>
#include <math.h>

typedef _Float16 h8v  __attribute__((ext_vector_type(8)));
typedef _Float16 v16h __attribute__((ext_vector_type(16)));
typedef float    v8f  __attribute__((ext_vector_type(8)));

#define QB 2048
#define QL 51
#define NLAY 13
#define PI_F 3.14159265358979323846f

// ---------------------------------------------------------------------------
// Kernel 1: per-row normalize + 51x51 encoder GEMM + tanh -> angles (B x 52)
// ---------------------------------------------------------------------------
__global__ void encode_kernel(const float* __restrict__ x,
                              const float* __restrict__ ew,
                              const float* __restrict__ eb,
                              float* __restrict__ ang) {
  __shared__ float row[64];
  __shared__ float red[64];
  __shared__ float s_mu, s_sd;
  const int b = blockIdx.x;
  const int t = threadIdx.x;              // 64 threads
  float v = (t < QL) ? x[(size_t)b * QL + t] : 0.f;
  red[t] = v;
  __syncthreads();
  #pragma unroll
  for (int o = 32; o > 0; o >>= 1) { if (t < o) red[t] += red[t + o]; __syncthreads(); }
  if (t == 0) s_mu = red[0] * (1.f / QL);
  __syncthreads();
  const float mu = s_mu;
  const float d = (t < QL) ? (v - mu) : 0.f;
  red[t] = d * d;
  __syncthreads();
  #pragma unroll
  for (int o = 32; o > 0; o >>= 1) { if (t < o) red[t] += red[t + o]; __syncthreads(); }
  if (t == 0) s_sd = sqrtf(red[0] * (1.f / QL));
  __syncthreads();
  const float sd = s_sd;
  row[t] = d / (sd + 1e-8f);              // xn (zeros for t>=51)
  __syncthreads();
  if (t < 52) {
    float acc = 0.f;
    if (t < QL) {
      acc = eb[t];
      for (int k = 0; k < QL; ++k) acc += row[k] * ew[k * QL + t];
      acc = tanhf(acc) * PI_F;
    }
    ang[(size_t)b * 52 + t] = acc;
  }
}

// ---------------------------------------------------------------------------
// Kernel 2: 8-qubit statevector circuit, one wave32 per batch element.
// s = b0..b7 (qubit0 = MSB). lane = s>>3 (qubits 0..4), reg r = s&7 (5..7).
// Adjacent gates fused: Rot*RY and Rot*RZ(reupload) on data qubits -> one
// cross-lane 2x2 apply each (13 shuffled gate-apps/layer instead of 17).
// ---------------------------------------------------------------------------
__device__ __forceinline__ void apply1q(float (&re)[8], float (&im)[8], int q, int lane,
                                        float u00r, float u00i, float u01r, float u01i,
                                        float u10r, float u10i, float u11r, float u11i) {
  if (q < 5) {
    const int lb = 4 - q, mask = 1 << lb;
    const int myBit = (lane >> lb) & 1;
    #pragma unroll
    for (int r = 0; r < 8; ++r) {
      float mr = re[r], mi = im[r];
      float pr = __shfl_xor(mr, mask, 32);
      float pi = __shfl_xor(mi, mask, 32);
      float a0r = myBit ? pr : mr, a0i = myBit ? pi : mi;
      float a1r = myBit ? mr : pr, a1i = myBit ? mi : pi;
      float w0r = myBit ? u10r : u00r, w0i = myBit ? u10i : u00i;
      float w1r = myBit ? u11r : u01r, w1i = myBit ? u11i : u01i;
      re[r] = w0r * a0r - w0i * a0i + w1r * a1r - w1i * a1i;
      im[r] = w0r * a0i + w0i * a0r + w1r * a1i + w1i * a1r;
    }
  } else {
    const int bit = 1 << (7 - q);
    #pragma unroll
    for (int r0 = 0; r0 < 8; ++r0) {
      if (r0 & bit) continue;
      const int r1 = r0 | bit;
      float ar = re[r0], ai = im[r0], br = re[r1], bi = im[r1];
      re[r0] = u00r * ar - u00i * ai + u01r * br - u01i * bi;
      im[r0] = u00r * ai + u00i * ar + u01r * bi + u01i * br;
      re[r1] = u10r * ar - u10i * ai + u11r * br - u11i * bi;
      im[r1] = u10r * ai + u10i * ar + u11r * bi + u11i * br;
    }
  }
}

struct Rmat { float r00, i00, r01, i01, r10, i10, r11, i11; };

// Rz(om)*Ry(th)*Rz(phi)
__device__ __forceinline__ Rmat rot_mat(const float* __restrict__ w) {
  const float phi = w[0], th = w[1], om = w[2];
  const float ct = __cosf(th * 0.5f), st = __sinf(th * 0.5f);
  const float al = 0.5f * (phi + om), be = 0.5f * (phi - om);
  const float ca = __cosf(al), sa = __sinf(al);
  const float cb = __cosf(be), sb = __sinf(be);
  Rmat m;
  m.r00 =  ca * ct; m.i00 = -sa * ct;
  m.r01 = -cb * st; m.i01 = -sb * st;
  m.r10 =  cb * st; m.i10 = -sb * st;
  m.r11 =  ca * ct; m.i11 =  sa * ct;
  return m;
}

// U = R * RY(c,s)
__device__ __forceinline__ void apply_rot_ry(float (&re)[8], float (&im)[8], int q, int lane,
                                             const Rmat& m, float c, float s) {
  apply1q(re, im, q, lane,
          m.r00 * c + m.r01 * s,  m.i00 * c + m.i01 * s,
         -m.r00 * s + m.r01 * c, -m.i00 * s + m.i01 * c,
          m.r10 * c + m.r11 * s,  m.i10 * c + m.i11 * s,
         -m.r10 * s + m.r11 * c, -m.i10 * s + m.i11 * c);
}

// U = R * diag(cz - i sz, cz + i sz)
__device__ __forceinline__ void apply_rot_rz(float (&re)[8], float (&im)[8], int q, int lane,
                                             const Rmat& m, float cz, float sz) {
  apply1q(re, im, q, lane,
          m.r00 * cz + m.i00 * sz,  m.i00 * cz - m.r00 * sz,
          m.r01 * cz - m.i01 * sz,  m.i01 * cz + m.r01 * sz,
          m.r10 * cz + m.i10 * sz,  m.i10 * cz - m.r10 * sz,
          m.r11 * cz - m.i11 * sz,  m.i11 * cz + m.r11 * sz);
}

// factor = cos(t/2) -/+ i*sin(t/2) by parity of (s & maskBits)
__device__ __forceinline__ void diag_phase(float (&re)[8], float (&im)[8], int lane,
                                           float c, float s, unsigned maskBits) {
  #pragma unroll
  for (int r = 0; r < 8; ++r) {
    const unsigned sidx = ((unsigned)lane << 3) | (unsigned)r;
    const int par = __popc(sidx & maskBits) & 1;
    const float fi = par ? s : -s;
    const float nr = re[r] * c - im[r] * fi;
    im[r] = re[r] * fi + im[r] * c;
    re[r] = nr;
  }
}

__device__ __forceinline__ void cry_gate(float (&re)[8], float (&im)[8], int lane,
                                         int qc, int qt, float c, float s) {
  const bool ctrl = ((lane >> (4 - qc)) & 1) != 0;
  if (qt == 4) {
    const int myBit = lane & 1;
    #pragma unroll
    for (int r = 0; r < 8; ++r) {
      float mr = re[r], mi = im[r];
      float pr = __shfl_xor(mr, 1, 32);
      float pi = __shfl_xor(mi, 1, 32);
      if (ctrl) {
        if (myBit == 0) { re[r] = c * mr - s * pr; im[r] = c * mi - s * pi; }
        else            { re[r] = s * pr + c * mr; im[r] = s * pi + c * mi; }
      }
    }
  } else {
    const int bit = 1 << (7 - qt);
    if (ctrl) {
      #pragma unroll
      for (int r0 = 0; r0 < 8; ++r0) {
        if (r0 & bit) continue;
        const int r1 = r0 | bit;
        float ar = re[r0], ai = im[r0], br = re[r1], bi = im[r1];
        re[r0] = c * ar - s * br;  im[r0] = c * ai - s * bi;
        re[r1] = s * ar + c * br;  im[r1] = s * ai + c * bi;
      }
    }
  }
}

__global__ void circuit_kernel(const float* __restrict__ angles,   // B x 52
                               const float* __restrict__ qw,       // 13x8x3
                               const float* __restrict__ qphi,     // 13x4x4
                               const float* __restrict__ qre,      // 13x4
                               const float* __restrict__ qde,      // 13x4
                               const float* __restrict__ qme,      // 13x4
                               _Float16* __restrict__ qA) {        // B x 64 f16
  const int lane = threadIdx.x & 31;
  const int b = blockIdx.x * (blockDim.x >> 5) + (threadIdx.x >> 5);
  float re[8], im[8];
  #pragma unroll
  for (int r = 0; r < 8; ++r) { re[r] = 0.f; im[r] = 0.f; }
  re[0] = (lane == 0) ? 1.f : 0.f;

  for (int l = 0; l < NLAY; ++l) {
    float ch[4];
    #pragma unroll
    for (int i = 0; i < 4; ++i) ch[i] = angles[(size_t)b * 52 + l * 4 + i];

    // fused data-RY + trainable Rot on qubits 0..3 (Rot matrices cached)
    Rmat Rm[4];
    #pragma unroll
    for (int i = 0; i < 4; ++i) {
      Rm[i] = rot_mat(qw + ((size_t)l * 8 + i) * 3);
      const float c = __cosf(ch[i] * 0.5f), s = __sinf(ch[i] * 0.5f);
      apply_rot_ry(re, im, i, lane, Rm[i], c, s);
    }
    // trainable Rot on qubits 4..7
    #pragma unroll
    for (int q = 4; q < 8; ++q) {
      const Rmat m = rot_mat(qw + ((size_t)l * 8 + q) * 3);
      apply1q(re, im, q, lane, m.r00, m.i00, m.r01, m.i01, m.r10, m.i10, m.r11, m.i11);
    }
    // data-entangle ZZ on pairs (0,1),(1,2),(2,3),(0,2)
    {
      const int pa[4] = {0, 1, 2, 0}, pb[4] = {1, 2, 3, 2};
      #pragma unroll
      for (int p = 0; p < 4; ++p) {
        const float t = qde[l * 4 + p];
        const float c = __cosf(t * 0.5f), s = __sinf(t * 0.5f);
        diag_phase(re, im, lane, c, s, (1u << (7 - pa[p])) | (1u << (7 - pb[p])));
      }
    }
    // fused reupload-RZ + second Rot on qubits 0..3 (reuses cached Rot)
    #pragma unroll
    for (int i = 0; i < 4; ++i) {
      const float t = ch[i] * qre[l * 4 + i];
      const float cz = __cosf(t * 0.5f), sz = __sinf(t * 0.5f);
      apply_rot_rz(re, im, i, lane, Rm[i], cz, sz);
    }
    // CRY(phi[l,i,j]) control i (0..3), target 4+j
    #pragma unroll
    for (int i = 0; i < 4; ++i) {
      #pragma unroll
      for (int j = 0; j < 4; ++j) {
        const float t = qphi[((size_t)l * 4 + i) * 4 + j];
        const float c = __cosf(t * 0.5f), s = __sinf(t * 0.5f);
        cry_gate(re, im, lane, i, 4 + j, c, s);
      }
    }
    // memory ZZ on (4,5),(5,6),(6,7),(7,4)
    #pragma unroll
    for (int i = 0; i < 4; ++i) {
      const int a = 4 + i, bq = 4 + ((i + 1) & 3);
      const float t = qme[l * 4 + i];
      const float c = __cosf(t * 0.5f), s = __sinf(t * 0.5f);
      diag_phase(re, im, lane, c, s, (1u << (7 - a)) | (1u << (7 - bq)));
    }
  }

  // ---- measurements ----
  float zv[8], zz[28], xr[8], xi[8];
  #pragma unroll
  for (int q = 0; q < 8; ++q) { zv[q] = 0.f; xr[q] = 0.f; xi[q] = 0.f; }
  #pragma unroll
  for (int k = 0; k < 28; ++k) zz[k] = 0.f;

  #pragma unroll
  for (int r = 0; r < 8; ++r) {
    const unsigned s = ((unsigned)lane << 3) | (unsigned)r;
    const float p = re[r] * re[r] + im[r] * im[r];
    float sg[8];
    #pragma unroll
    for (int q = 0; q < 8; ++q) sg[q] = ((s >> (7 - q)) & 1) ? -p : p;
    #pragma unroll
    for (int q = 0; q < 8; ++q) zv[q] += sg[q];
    int k = 0;
    #pragma unroll
    for (int a = 0; a < 8; ++a) {
      #pragma unroll
      for (int bq = a + 1; bq < 8; ++bq) {
        zz[k] += ((s >> (7 - bq)) & 1) ? -sg[a] : sg[a];
        ++k;
      }
    }
  }
  // <X>,<Y>: c = sum_{bit=0} conj(psi0)*psi1
  #pragma unroll
  for (int q = 0; q < 5; ++q) {
    const int lb = 4 - q, mask = 1 << lb;
    const int myBit = (lane >> lb) & 1;
    #pragma unroll
    for (int r = 0; r < 8; ++r) {
      float pr = __shfl_xor(re[r], mask, 32);
      float pi = __shfl_xor(im[r], mask, 32);
      if (!myBit) {
        xr[q] += re[r] * pr + im[r] * pi;
        xi[q] += re[r] * pi - im[r] * pr;
      }
    }
  }
  #pragma unroll
  for (int q = 5; q < 8; ++q) {
    const int bit = 1 << (7 - q);
    #pragma unroll
    for (int r0 = 0; r0 < 8; ++r0) {
      if (r0 & bit) continue;
      const int r1 = r0 | bit;
      xr[q] += re[r0] * re[r1] + im[r0] * im[r1];
      xi[q] += re[r0] * im[r1] - im[r0] * re[r1];
    }
  }

  auto wred = [](float v) {
    #pragma unroll
    for (int o = 16; o > 0; o >>= 1) v += __shfl_xor(v, o, 32);
    return v;
  };

  float feat[52];
  #pragma unroll
  for (int q = 0; q < 8; ++q) {
    feat[3 * q + 0] = 2.f * wred(xr[q]);
    feat[3 * q + 1] = 2.f * wred(xi[q]);
    feat[3 * q + 2] = wred(zv[q]);
  }
  #pragma unroll
  for (int k = 0; k < 28; ++k) feat[24 + k] = wred(zz[k]);

  if (lane == 0) {
    _Float16* out = qA + (size_t)b * 64;
    #pragma unroll
    for (int i = 0; i < 52; ++i) out[i] = (_Float16)feat[i];
    #pragma unroll
    for (int i = 52; i < 64; ++i) out[i] = (_Float16)0.f;
  }
}

// ---------------------------------------------------------------------------
// Kernel 3: repack f32 weight (Kreal x N, zero-pad to Kpad) into WMMA
// B-fragment order: dst[((kt*NT+nt)*32 + lane)*16 + e], k = kt*32+e+16*(lane>>4),
// n = nt*16 + (lane&15). GEMM B loads become two contiguous b128 per lane.
// ---------------------------------------------------------------------------
__global__ void pack_b_kernel(const float* __restrict__ src, _Float16* __restrict__ dst,
                              int Kreal, int Kpad, int N,
                              size_t srcHeadStride, size_t dstHeadStride) {
  const int h = blockIdx.z;
  src += (size_t)h * srcHeadStride;
  dst += (size_t)h * dstHeadStride;
  const int total = Kpad * N;
  const int idx = blockIdx.x * blockDim.x + threadIdx.x;
  if (idx >= total) return;
  const int e = idx & 15;
  const int l = (idx >> 4) & 31;
  const int t = idx >> 9;                 // kt*NT + nt
  const int NT = N >> 4;
  const int nt = t % NT;
  const int kt = t / NT;
  const int n = (nt << 4) + (l & 15);
  const int k = (kt << 5) + e + 16 * (l >> 4);
  const float v = (k < Kreal) ? src[(size_t)k * N + n] : 0.f;
  dst[idx] = (_Float16)v;
}

// ---------------------------------------------------------------------------
// Kernel 4: WMMA GEMM  C = elu(A(MxK f16 row-major) * Bpacked(KxN) + bias).
// One 32(M) x 64(N) tile per wave: 2 A-frags x 4 B-frags -> 8 wmma per
// 32-deep K step (1.5 b128 loads per wmma). 4 waves/block along M.
// grid = (N/64, M/128, heads).
// ---------------------------------------------------------------------------
__global__ void wmma_gemm_elu(const _Float16* __restrict__ A,
                              const _Float16* __restrict__ Bp,
                              const float* __restrict__ bias,
                              _Float16* __restrict__ C,
                              int M, int N, int K,
                              size_t aHeadStride, size_t bHeadStride,
                              size_t biasHeadStride, size_t cHeadStride) {
  const int head = blockIdx.z;
  A    += (size_t)head * aHeadStride;
  Bp   += (size_t)head * bHeadStride;
  bias += (size_t)head * biasHeadStride;
  C    += (size_t)head * cHeadStride;

  const int lane = threadIdx.x & 31;
  const int wave = threadIdx.x >> 5;
  const int lr   = lane & 15;
  const int g    = lane >> 4;
  const int mbase = (blockIdx.y * 4 + wave) * 32;
  const int nbase = blockIdx.x * 64;
  const int NT = N >> 4;
  const int nt0 = nbase >> 4;

  const _Float16* arow0 = A + (size_t)(mbase + lr) * K;
  const _Float16* arow1 = arow0 + (size_t)16 * K;

  v8f acc[2][4] = {};
  const int KT = K >> 5;
  for (int kt = 0; kt < KT; ++kt) {
    v16h af[2], bf[4];
    {
      // A fragment: k = e+16g (contiguous, matches packed B)
      h8v lo0 = *(const h8v*)(arow0 + kt * 32 + 16 * g);
      h8v hi0 = *(const h8v*)(arow0 + kt * 32 + 8 + 16 * g);
      h8v lo1 = *(const h8v*)(arow1 + kt * 32 + 16 * g);
      h8v hi1 = *(const h8v*)(arow1 + kt * 32 + 8 + 16 * g);
      #pragma unroll
      for (int i = 0; i < 8; ++i) {
        af[0][i] = lo0[i]; af[0][i + 8] = hi0[i];
        af[1][i] = lo1[i]; af[1][i + 8] = hi1[i];
      }
    }
    #pragma unroll
    for (int nj = 0; nj < 4; ++nj) {
      const _Float16* bp = Bp + (((size_t)kt * NT + nt0 + nj) * 32 + lane) * 16;
      h8v blo = *(const h8v*)(bp);
      h8v bhi = *(const h8v*)(bp + 8);
      #pragma unroll
      for (int i = 0; i < 8; ++i) { bf[nj][i] = blo[i]; bf[nj][i + 8] = bhi[i]; }
    }
    #pragma unroll
    for (int mi = 0; mi < 2; ++mi) {
      #pragma unroll
      for (int nj = 0; nj < 4; ++nj) {
        acc[mi][nj] = __builtin_amdgcn_wmma_f32_16x16x32_f16(
            false, af[mi], false, bf[nj], (short)0, acc[mi][nj], false, false);
      }
    }
  }

  #pragma unroll
  for (int nj = 0; nj < 4; ++nj) {
    const int ncol = nbase + nj * 16 + lr;
    const float bv = bias[ncol];
    #pragma unroll
    for (int mi = 0; mi < 2; ++mi) {
      #pragma unroll
      for (int e = 0; e < 8; ++e) {
        const int row = mbase + mi * 16 + e + 8 * g;   // D: VGPR e -> M = e + 8*(lane>>4)
        float v = acc[mi][nj][e] + bv;
        v = v > 0.f ? v : (__expf(v) - 1.f);           // ELU
        C[(size_t)row * N + ncol] = (_Float16)v;
      }
    }
  }
}

// ---------------------------------------------------------------------------
// Kernel 5: final 64->2 head projection; mus + softplus(sigmas).
// d_out = mus (B*7) then sigmas2 (B*7), f32.
// ---------------------------------------------------------------------------
__global__ void head_out_kernel(const _Float16* __restrict__ o1,  // 7 x B x 64
                                const float* __restrict__ w2,     // 7 x 64 x 2
                                const float* __restrict__ b2,     // 7 x 2
                                float* __restrict__ out) {
  const int idx = blockIdx.x * blockDim.x + threadIdx.x;  // b*7+h
  if (idx >= QB * 7) return;
  const int h = idx % 7, b = idx / 7;
  const _Float16* row = o1 + ((size_t)h * QB + b) * 64;
  const float* w = w2 + (size_t)h * 128;
  float m = b2[h * 2 + 0], sg = b2[h * 2 + 1];
  #pragma unroll 8
  for (int k = 0; k < 64; ++k) {
    const float v = (float)row[k];
    m  += v * w[k * 2 + 0];
    sg += v * w[k * 2 + 1];
  }
  out[idx] = m;
  out[QB * 7 + idx] = (sg > 20.f) ? sg : log1pf(__expf(sg));
}

// ---------------------------------------------------------------------------
extern "C" void kernel_launch(void* const* d_in, const int* in_sizes, int n_in,
                              void* d_out, int out_size, void* d_ws, size_t ws_size,
                              hipStream_t stream) {
  const float* x      = (const float*)d_in[0];   // (2048, 51)
  const float* enc_w  = (const float*)d_in[1];   // (51, 51)
  const float* enc_b  = (const float*)d_in[2];   // (51,)
  const float* q_w    = (const float*)d_in[3];   // (13, 8, 3)
  const float* q_phi  = (const float*)d_in[4];   // (13, 4, 4)
  const float* q_re   = (const float*)d_in[5];   // (13, 4)
  const float* q_de   = (const float*)d_in[6];   // (13, 4)
  const float* q_me   = (const float*)d_in[7];   // (13, 4)
  const float* lin0_w = (const float*)d_in[8];   // (52, 512)
  const float* lin0_b = (const float*)d_in[9];
  const float* lin1_w = (const float*)d_in[10];  // (512, 512)
  const float* lin1_b = (const float*)d_in[11];
  const float* linl_w = (const float*)d_in[12];  // (512, 128)
  const float* linl_b = (const float*)d_in[13];
  const float* hw0    = (const float*)d_in[14];  // (7, 128, 128)
  const float* hb0    = (const float*)d_in[15];  // (7, 128)
  const float* hw1    = (const float*)d_in[16];  // (7, 128, 64)
  const float* hb1    = (const float*)d_in[17];  // (7, 64)
  const float* hw2    = (const float*)d_in[18];  // (7, 64, 2)
  const float* hb2    = (const float*)d_in[19];  // (7, 2)
  float* out = (float*)d_out;

  size_t off = 0;
  auto wsalloc = [&](size_t bytes) -> void* {
    void* p = (char*)d_ws + off;
    off += (bytes + 255) & ~(size_t)255;
    return p;
  };
  float*    angles = (float*)   wsalloc((size_t)QB * 52 * 4);
  _Float16* qA     = (_Float16*)wsalloc((size_t)QB * 64 * 2);
  _Float16* pB0    = (_Float16*)wsalloc((size_t)64 * 512 * 2);
  _Float16* pB1    = (_Float16*)wsalloc((size_t)512 * 512 * 2);
  _Float16* pBL    = (_Float16*)wsalloc((size_t)512 * 128 * 2);
  _Float16* pH0    = (_Float16*)wsalloc((size_t)7 * 128 * 128 * 2);
  _Float16* pH1    = (_Float16*)wsalloc((size_t)7 * 128 * 64 * 2);
  _Float16* h0     = (_Float16*)wsalloc((size_t)QB * 512 * 2);
  _Float16* h1     = (_Float16*)wsalloc((size_t)QB * 512 * 2);
  _Float16* hL     = (_Float16*)wsalloc((size_t)QB * 128 * 2);
  _Float16* o0     = (_Float16*)wsalloc((size_t)7 * QB * 128 * 2);
  _Float16* o1     = (_Float16*)wsalloc((size_t)7 * QB * 64 * 2);

  // 1) encoder
  encode_kernel<<<QB, 64, 0, stream>>>(x, enc_w, enc_b, angles);

  // 2) quantum circuit, one wave32 per sample (8 waves/block)
  circuit_kernel<<<QB / 8, 256, 0, stream>>>(angles, q_w, q_phi, q_re, q_de, q_me, qA);

  // 3) weight repack into WMMA B-fragment layout
  {
    int n;
    n = 64 * 512;
    pack_b_kernel<<<dim3((n + 255) / 256, 1, 1), 256, 0, stream>>>(lin0_w, pB0, 52, 64, 512, 0, 0);
    n = 512 * 512;
    pack_b_kernel<<<dim3((n + 255) / 256, 1, 1), 256, 0, stream>>>(lin1_w, pB1, 512, 512, 512, 0, 0);
    n = 512 * 128;
    pack_b_kernel<<<dim3((n + 255) / 256, 1, 1), 256, 0, stream>>>(linl_w, pBL, 512, 512, 128, 0, 0);
    n = 128 * 128;
    pack_b_kernel<<<dim3((n + 255) / 256, 1, 7), 256, 0, stream>>>(hw0, pH0, 128, 128, 128,
                                                                   (size_t)128 * 128, (size_t)128 * 128);
    n = 128 * 64;
    pack_b_kernel<<<dim3((n + 255) / 256, 1, 7), 256, 0, stream>>>(hw1, pH1, 128, 128, 64,
                                                                   (size_t)128 * 64, (size_t)128 * 64);
  }

  // 4) dense chain via WMMA: 32x64 tile/wave, 4 waves/block along M
  // qA (2048x64) * lin0 (64x512) -> h0
  wmma_gemm_elu<<<dim3(512 / 64, QB / 128, 1), 128, 0, stream>>>(
      qA, pB0, lin0_b, h0, QB, 512, 64, 0, 0, 0, 0);
  // h0 * lin1 (512x512) -> h1
  wmma_gemm_elu<<<dim3(512 / 64, QB / 128, 1), 128, 0, stream>>>(
      h0, pB1, lin1_b, h1, QB, 512, 512, 0, 0, 0, 0);
  // h1 * linl (512x128) -> hL
  wmma_gemm_elu<<<dim3(128 / 64, QB / 128, 1), 128, 0, stream>>>(
      h1, pBL, linl_b, hL, QB, 128, 512, 0, 0, 0, 0);
  // heads stage 0: hL (shared A) * hw0[h] (128x128) -> o0[h]
  wmma_gemm_elu<<<dim3(128 / 64, QB / 128, 7), 128, 0, stream>>>(
      hL, pH0, hb0, o0, QB, 128, 128,
      0, (size_t)128 * 128, 128, (size_t)QB * 128);
  // heads stage 1: o0[h] * hw1[h] (128x64) -> o1[h]
  wmma_gemm_elu<<<dim3(64 / 64, QB / 128, 7), 128, 0, stream>>>(
      o0, pH1, hb1, o1, QB, 64, 128,
      (size_t)QB * 128, (size_t)128 * 64, 64, (size_t)QB * 64);

  // 5) final 64->2 projection + softplus
  head_out_kernel<<<(QB * 7 + 255) / 256, 256, 0, stream>>>(o1, hw2, hb2, out);
}
